// SSA_5909875000108
// MI455X (gfx1250) — matrix operand order repaired and checked by reading, not compiled, not run on hardware
//
#include <hip/hip_runtime.h>
#include <hip/hip_bf16.h>

// Problem constants (match reference)
#define N_SEQ 8192
#define D_IN  1024
#define DKV   128      // d_k == d_v == 128

// Flash-attention tiling
#define KT    64       // keys per LDS tile
#define KSTR  136      // K-tile LDS row stride (bf16 elems), 272B -> bank skew
#define VSTR  72       // V-tile LDS row stride (bf16 elems), 144B -> bank skew
#define PSTR  72       // P staging stride

typedef __bf16 bf16_t;
typedef __attribute__((ext_vector_type(16))) __bf16 bf16x16;
typedef __attribute__((ext_vector_type(8)))  __bf16 bf16x8;
typedef __attribute__((ext_vector_type(8)))  float  f32x8;
typedef __attribute__((ext_vector_type(4)))  float  f32x4;

// Types for the async global->LDS builtin (per compiler diagnostic: pointer
// to 16B int vector; AS1 source, AS3 destination).
typedef int v4i_t __attribute__((vector_size(16)));
typedef __attribute__((address_space(1))) v4i_t* g_v4i_p;
typedef __attribute__((address_space(3))) v4i_t* l_v4i_p;

#define NEG_BIG (-3.0e38f)

__device__ __forceinline__ bf16x16 join8(bf16x8 lo, bf16x8 hi) {
  bf16x16 r;
#pragma unroll
  for (int i = 0; i < 8; ++i) { r[i] = lo[i]; r[i + 8] = hi[i]; }
  return r;
}

// D = A(16x32) x B(32x16) + C, bf16 inputs, f32 accumulate
__device__ __forceinline__ f32x8 wmma_bf16(bf16x16 a, bf16x16 b, f32x8 c) {
  return __builtin_amdgcn_wmma_f32_16x16x32_bf16(
      /*neg_a=*/false, a, /*neg_b=*/false, b,
      /*c_mod=*/(short)0, c, /*reuse_a=*/false, /*reuse_b=*/false);
}

// Async 16B global -> LDS copy (per-lane addresses), tracked by ASYNCcnt.
__device__ __forceinline__ void async_cp16(const bf16_t* g, bf16_t* l) {
#if __has_builtin(__builtin_amdgcn_global_load_async_to_lds_b128)
  __builtin_amdgcn_global_load_async_to_lds_b128((g_v4i_p)g, (l_v4i_p)l, 0, 0);
#else
  unsigned lofs = (unsigned)(size_t)((l_v4i_p)l);
  asm volatile("global_load_async_to_lds_b128 %0, %1, off"
               :: "v"(lofs), "v"(g) : "memory");
#endif
}

__device__ __forceinline__ void wait_async_le8() {
#if __has_builtin(__builtin_amdgcn_s_wait_asynccnt)
  __builtin_amdgcn_s_wait_asynccnt(8);
#else
  asm volatile("s_wait_asynccnt 0x8" ::: "memory");
#endif
}
__device__ __forceinline__ void wait_async_0() {
#if __has_builtin(__builtin_amdgcn_s_wait_asynccnt)
  __builtin_amdgcn_s_wait_asynccnt(0);
#else
  asm volatile("s_wait_asynccnt 0x0" ::: "memory");
#endif
}

// A fragment (16x32, MxK) from a row-major bf16 matrix [rows][ld].
// Lane l<16 : M=l,    K in {0..15}
// Lane l>=16: M=l-16, K in {16..31}
__device__ __forceinline__ bf16x16 loadA_bf16(const bf16_t* base, int ld,
                                              int row0, int k0, int lane) {
  const int lp = lane & 15, h = lane >> 4;
  const bf16_t* p = base + (size_t)(row0 + lp) * ld + k0 + h * 16;
  bf16x8 lo = *(const bf16x8*)(p);
  bf16x8 hi = *(const bf16x8*)(p + 8);
  return join8(lo, hi);
}

// A fragment from a row-major f32 matrix, converted to bf16 on the fly.
__device__ __forceinline__ bf16x16 loadA_f32(const float* base, int ld,
                                             int row0, int k0, int lane) {
  const int lp = lane & 15, h = lane >> 4;
  const float* p = base + (size_t)(row0 + lp) * ld + k0 + h * 16;
  f32x4 a0 = *(const f32x4*)(p);
  f32x4 a1 = *(const f32x4*)(p + 4);
  f32x4 b0 = *(const f32x4*)(p + 8);
  f32x4 b1 = *(const f32x4*)(p + 12);
  bf16x16 r;
#pragma unroll
  for (int i = 0; i < 4; ++i) {
    r[i]      = (__bf16)a0[i];
    r[i + 4]  = (__bf16)a1[i];
    r[i + 8]  = (__bf16)b0[i];
    r[i + 12] = (__bf16)b1[i];
  }
  return r;
}

// B fragment (32x16, KxN) where B[k][n] = src[n0+n][k0+k] for a row-major
// bf16 matrix src[*][ld] (src rows become B columns; K contiguous per lane).
// Lane l<16 : N=l,    K=0..15   Lane l>=16: N=l-16, K=16..31
__device__ __forceinline__ bf16x16 loadB_bf16(const bf16_t* base, long ld,
                                              int n0, int k0, int lane) {
  const int lp = lane & 15, h = lane >> 4;
  const bf16_t* p = base + (size_t)(n0 + lp) * ld + k0 + h * 16;
  bf16x8 lo = *(const bf16x8*)(p);
  bf16x8 hi = *(const bf16x8*)(p + 8);
  return join8(lo, hi);
}

// ---------------------------------------------------------------------------
// Kernel 1: transpose+convert weights: Wt[mat][n][k] = W[mat][k][n] (bf16)
// ---------------------------------------------------------------------------
__global__ void ssa_prep_weights(const float* __restrict__ Wq,
                                 const float* __restrict__ Wk,
                                 const float* __restrict__ Wv,
                                 bf16_t* __restrict__ Wt) {
  int idx = blockIdx.x * blockDim.x + threadIdx.x;
  if (idx >= 3 * DKV * D_IN) return;
  int mat = idx / (DKV * D_IN);
  int rem = idx % (DKV * D_IN);
  int n = rem / D_IN;
  int k = rem % D_IN;
  const float* W = (mat == 0) ? Wq : ((mat == 1) ? Wk : Wv);
  Wt[idx] = (__bf16)W[(size_t)k * DKV + n];
}

// ---------------------------------------------------------------------------
// Kernel 2: fused QKV projection. Block=256 (8 waves) handles 16 rows of X.
// 24 jobs per block = {Q,K,V} x 8 column tiles of 16.
// Outputs: Qb (row-major bf16, pre-scaled 1/sqrt(dk)), Kb (row-major bf16),
//          Vt (transposed bf16: Vt[n][key]).
// ---------------------------------------------------------------------------
__global__ __launch_bounds__(256) void ssa_qkv_proj(
    const float* __restrict__ X, const bf16_t* __restrict__ Wt,
    const float* __restrict__ bq, const float* __restrict__ bk,
    const float* __restrict__ bv, bf16_t* __restrict__ Qb,
    bf16_t* __restrict__ Kb, bf16_t* __restrict__ Vt) {
  const int lane = threadIdx.x & 31;
  const int wave = threadIdx.x >> 5;
  const int row0 = blockIdx.x * 16;
  const int lp = lane & 15, h = lane >> 4;

  for (int job = wave; job < 24; job += 8) {
    const int mat = job >> 3;  // 0=Q 1=K 2=V
    const int nt  = job & 7;   // 16-wide column tile
    const bf16_t* Wmat = Wt + (size_t)mat * DKV * D_IN;

    f32x8 acc = {};
#pragma unroll 4
    for (int k0 = 0; k0 < D_IN; k0 += 32) {
      bf16x16 a = loadA_f32(X, D_IN, row0, k0, lane);
      bf16x16 b = loadB_bf16(Wmat, D_IN, nt * 16, k0, lane);
      acc = wmma_bf16(a, b, acc);
    }

    const float* bias = (mat == 0) ? bq : ((mat == 1) ? bk : bv);
    const float bcol = bias[nt * 16 + lp];

    if (mat == 0) {
      const float scale = 0.088388347648318447f;  // 1/sqrt(128)
#pragma unroll
      for (int r = 0; r < 8; ++r) {
        int row = row0 + h * 8 + r;
        Qb[(size_t)row * DKV + nt * 16 + lp] = (__bf16)((acc[r] + bcol) * scale);
      }
    } else if (mat == 1) {
#pragma unroll
      for (int r = 0; r < 8; ++r) {
        int row = row0 + h * 8 + r;
        Kb[(size_t)row * DKV + nt * 16 + lp] = (__bf16)(acc[r] + bcol);
      }
    } else {
      bf16x8 pack;
#pragma unroll
      for (int r = 0; r < 8; ++r) pack[r] = (__bf16)(acc[r] + bcol);
      // Vt[n][key]: keys row0+8h .. row0+8h+7 contiguous -> one 16B store
      *(bf16x8*)(Vt + (size_t)(nt * 16 + lp) * N_SEQ + row0 + h * 8) = pack;
    }
  }
}

// ---------------------------------------------------------------------------
// Cooperative async fill of one 64-key K/V tile into LDS (8 ops/thread).
// K tile: [64][KSTR] row-major (keys x dk), V tile: [128][VSTR] (dv x keys).
// ---------------------------------------------------------------------------
__device__ __forceinline__ void issue_kv_tile(const bf16_t* __restrict__ Kb,
                                              const bf16_t* __restrict__ Vt,
                                              bf16_t* ksh, bf16_t* vsh,
                                              int kb, int tid) {
#pragma unroll
  for (int i = 0; i < 4; ++i) {           // K: 64 rows x 256B = 1024 chunks
    int c = tid + i * 256;
    int row = c >> 4, col = c & 15;
    async_cp16(Kb + (size_t)(kb + row) * DKV + col * 8,
               ksh + row * KSTR + col * 8);
  }
#pragma unroll
  for (int i = 0; i < 4; ++i) {           // V: 128 rows x 128B = 1024 chunks
    int c = tid + i * 256;
    int n = c >> 3, col = c & 7;
    async_cp16(Vt + (size_t)n * N_SEQ + kb + col * 8,
               vsh + n * VSTR + col * 8);
  }
}

// ---------------------------------------------------------------------------
// Kernel 3: causal flash attention. Block = 8 waves = 128 query rows.
// K/V tiles (64 keys) are double-buffered in LDS via async global->LDS loads
// and shared by all 8 waves. Online softmax over 64-key tiles.
// ---------------------------------------------------------------------------
__global__ __launch_bounds__(256) void ssa_flash_attn(
    const bf16_t* __restrict__ Qb, const bf16_t* __restrict__ Kb,
    const bf16_t* __restrict__ Vt, float* __restrict__ Out) {
  __shared__ __align__(16) bf16_t Ksh[2][KT * KSTR];    // 2 x 17408B
  __shared__ __align__(16) bf16_t Vsh[2][DKV * VSTR];   // 2 x 18432B
  __shared__ __align__(16) bf16_t Plds[8][16 * PSTR];   // 18432B

  const int tid  = threadIdx.x;
  const int lane = tid & 31;
  const int wave = tid >> 5;
  const int lp = lane & 15, h = lane >> 4;
  const int q0    = blockIdx.x * 128;
  const int qbase = q0 + wave * 16;

  // Q fragments for this wave's 16x128 query tile (4 K-steps of 32)
  bf16x16 aQ[4];
#pragma unroll
  for (int kk = 0; kk < 4; ++kk)
    aQ[kk] = loadA_bf16(Qb, DKV, qbase, kk * 32, lane);

  f32x8 O[8] = {};
  float m_r[8], l_r[8];
#pragma unroll
  for (int r = 0; r < 8; ++r) { m_r[r] = NEG_BIG; l_r[r] = 0.0f; }

  bf16_t* pw = &Plds[wave][0];

  const int ntiles = (q0 + 128) / KT;  // keys 0 .. q0+127

  issue_kv_tile(Kb, Vt, Ksh[0], Vsh[0], 0, tid);

  for (int jt = 0; jt < ntiles; ++jt) {
    const int kb  = jt * KT;
    const int buf = jt & 1;

    __syncthreads();  // consumers of tile jt-1 (parity buf^1) are done
    if (jt + 1 < ntiles) {
      issue_kv_tile(Kb, Vt, Ksh[buf ^ 1], Vsh[buf ^ 1], kb + KT, tid);
      wait_async_le8();   // oldest 8 (tile jt) complete; next tile in flight
    } else {
      wait_async_0();
    }
    __syncthreads();  // tile jt visible to all waves

    if (kb <= qbase + 15) {  // wave-uniform causal skip
      const bf16_t* ksh = Ksh[buf];
      const bf16_t* vsh = Vsh[buf];

      // ---- S = Q K^T : four 16x16 C fragments over 64 keys ----
      f32x8 S[4] = {};
#pragma unroll
      for (int kk = 0; kk < 4; ++kk) {
#pragma unroll
        for (int s = 0; s < 4; ++s) {
          bf16x16 bK = loadB_bf16(ksh, KSTR, s * 16, kk * 32, lane);
          S[s] = wmma_bf16(aQ[kk], bK, S[s]);
        }
      }

      // ---- causal mask ----
      if (kb + KT - 1 > qbase) {
#pragma unroll
        for (int s = 0; s < 4; ++s) {
          const int col = kb + s * 16 + lp;
#pragma unroll
          for (int r = 0; r < 8; ++r) {
            const int row = qbase + h * 8 + r;
            if (col > row) S[s][r] = NEG_BIG;
          }
        }
      }

      // ---- online softmax over 64 columns ----
#pragma unroll
      for (int r = 0; r < 8; ++r) {
        float v = fmaxf(fmaxf(S[0][r], S[1][r]), fmaxf(S[2][r], S[3][r]));
#pragma unroll
        for (int off = 1; off < 16; off <<= 1)
          v = fmaxf(v, __shfl_xor(v, off, 32));
        const float mnew  = fmaxf(m_r[r], v);
        const float alpha = __expf(m_r[r] - mnew);
        float rs = 0.0f;
#pragma unroll
        for (int s = 0; s < 4; ++s) {
          S[s][r] = __expf(S[s][r] - mnew);
          rs += S[s][r];
        }
#pragma unroll
        for (int off = 1; off < 16; off <<= 1)
          rs += __shfl_xor(rs, off, 32);
        l_r[r] = l_r[r] * alpha + rs;
        m_r[r] = mnew;
#pragma unroll
        for (int vt = 0; vt < 8; ++vt) O[vt][r] *= alpha;
      }

      // ---- stage P (16x64 bf16) in per-wave LDS: C-layout -> row-major ----
#pragma unroll
      for (int r = 0; r < 8; ++r) {
        const int row = h * 8 + r;
#pragma unroll
        for (int s = 0; s < 4; ++s)
          pw[row * PSTR + s * 16 + lp] = (__bf16)S[s][r];
      }
      asm volatile("s_wait_dscnt 0" ::: "memory");  // wave-local LDS RAW

      // ---- O += P * V : 2 K-steps x 8 d_v tiles ----
#pragma unroll
      for (int ks = 0; ks < 2; ++ks) {
        bf16x8 plo = *(const bf16x8*)(pw + lp * PSTR + ks * 32 + h * 16);
        bf16x8 phi = *(const bf16x8*)(pw + lp * PSTR + ks * 32 + h * 16 + 8);
        bf16x16 aP = join8(plo, phi);
#pragma unroll
        for (int vt = 0; vt < 8; ++vt) {
          bf16x16 bV = loadB_bf16(vsh, VSTR, vt * 16, ks * 32, lane);
          O[vt] = wmma_bf16(aP, bV, O[vt]);
        }
      }
    }
  }

  // ---- epilogue: normalize and store fp32 ----
#pragma unroll
  for (int r = 0; r < 8; ++r) {
    const float inv = 1.0f / l_r[r];
    const int row = qbase + h * 8 + r;
#pragma unroll
    for (int vt = 0; vt < 8; ++vt)
      Out[(size_t)row * DKV + vt * 16 + lp] = O[vt][r] * inv;
  }
}

// ---------------------------------------------------------------------------
extern "C" void kernel_launch(void* const* d_in, const int* in_sizes, int n_in,
                              void* d_out, int out_size, void* d_ws,
                              size_t ws_size, hipStream_t stream) {
  const float* X  = (const float*)d_in[0];
  const float* Wq = (const float*)d_in[1];
  const float* bq = (const float*)d_in[2];
  const float* Wk = (const float*)d_in[3];
  const float* bk = (const float*)d_in[4];
  const float* Wv = (const float*)d_in[5];
  const float* bv = (const float*)d_in[6];
  float* Out = (float*)d_out;

  char* ws = (char*)d_ws;
  bf16_t* Wt = (bf16_t*)ws;                                   // 3*128*1024 bf16
  bf16_t* Qb = (bf16_t*)(ws + (size_t)3 * DKV * D_IN * 2);    // [N][128] bf16
  bf16_t* Kb = Qb + (size_t)N_SEQ * DKV;                      // [N][128] bf16
  bf16_t* Vt = Kb + (size_t)N_SEQ * DKV;                      // [128][N] bf16

  ssa_prep_weights<<<(3 * DKV * D_IN + 255) / 256, 256, 0, stream>>>(Wq, Wk, Wv,
                                                                     Wt);
  ssa_qkv_proj<<<N_SEQ / 16, 256, 0, stream>>>(X, Wt, bq, bk, bv, Qb, Kb, Vt);
  ssa_flash_attn<<<N_SEQ / 128, 256, 0, stream>>>(Qb, Kb, Vt, Out);
}
